// CausalSelfAttention_41790031790233
// MI455X (gfx1250) — compile-verified, matrix operand order checked by reading, not run on hardware
//
#include <hip/hip_runtime.h>

// ---------------------------------------------------------------------------
// CDNA5 (gfx1250) fused attention pipeline, bf16 WMMA + async-to-LDS staging.
//   B=2, T=2048, D=2048, NH=16, HD=128  (all fixed by the reference)
// ---------------------------------------------------------------------------

typedef __bf16 bf16_t;
typedef __attribute__((ext_vector_type(16))) __bf16 v16bf;
typedef __attribute__((ext_vector_type(8)))  float  v8f;

__device__ __forceinline__ bf16_t to_bf16(float f) {
  unsigned u;
  __builtin_memcpy(&u, &f, 4);
  u += 0x7FFFu + ((u >> 16) & 1u);           // round-to-nearest-even
  unsigned short h = (unsigned short)(u >> 16);
  bf16_t b;
  __builtin_memcpy(&b, &h, 2);
  return b;
}

__device__ __forceinline__ v8f wmma_bf16(v16bf a, v16bf b, v8f c) {
  // v_wmma_f32_16x16x32_bf16: D = A(16x32) * B(32x16) + C(16x16), fp32 accum
  return __builtin_amdgcn_wmma_f32_16x16x32_bf16(
      /*neg_a=*/false, a, /*neg_b=*/false, b,
      /*c_mod=*/(short)0, c, /*reuse_a=*/false, /*reuse_b=*/false);
}

// CDNA5 async DMA: 16B per lane, global -> LDS, tracked by ASYNCcnt.
// LDS destination address (byte offset) in a VGPR, 64-bit global addr in pair.
__device__ __forceinline__ void async_b128(unsigned lds_off, const void* gaddr) {
  asm volatile("global_load_async_to_lds_b128 %0, %1, off"
               :: "v"(lds_off), "v"(gaddr)
               : "memory");
}
__device__ __forceinline__ void wait_async0() {
  asm volatile("s_wait_asynccnt 0x0" ::: "memory");
}
__device__ __forceinline__ unsigned lds_off_of(const void* p) {
  // generic pointer to __shared__: low 32 bits carry the LDS byte offset
  return (unsigned)(size_t)p;
}

// ---------------------------------------------------------------------------
// Elementwise fp32 -> bf16 cast (grid-stride)
// ---------------------------------------------------------------------------
__global__ void cast_bf16_kernel(const float* __restrict__ in,
                                 bf16_t* __restrict__ out, size_t n) {
  size_t i = (size_t)blockIdx.x * blockDim.x + threadIdx.x;
  size_t stride = (size_t)gridDim.x * blockDim.x;
  for (; i < n; i += stride) out[i] = to_bf16(in[i]);
}

// ---------------------------------------------------------------------------
// W[K][N] fp32  ->  Wt[N][K] bf16   (so WMMA B-fragments read K-contiguous)
// ---------------------------------------------------------------------------
__global__ void transpose_cast_kernel(const float* __restrict__ W,
                                      bf16_t* __restrict__ Wt, int K, int N) {
  size_t idx = (size_t)blockIdx.x * blockDim.x + threadIdx.x;
  size_t total = (size_t)K * N;
  if (idx >= total) return;
  int n = (int)(idx / K);
  int k = (int)(idx % K);
  Wt[idx] = to_bf16(W[(size_t)k * N + n]);   // write coalesced in k
}

// ---------------------------------------------------------------------------
// Tiled bf16 GEMM: C[M][N] = A[M][K] * Bt[N][K]^T, 128x128 block tile,
// 8 waves (wave tile 32x64), K staged 32-deep, DOUBLE-BUFFERED via
// global_load_async_to_lds_b128 (copy of tile k+1 overlaps WMMAs on tile k).
// mode 0: Q epilogue (RoPE, store [B,NH,T,HD] bf16)
// mode 1: K epilogue (RoPE, store [B,NH,T,HD] bf16)
// mode 2: V epilogue (store transposed [B,NH,HD,T] bf16)
// mode 3: output projection (store fp32 [M][N])
// ---------------------------------------------------------------------------
__global__ __launch_bounds__(256) void gemm_bf16_kernel(
    const bf16_t* __restrict__ A, const bf16_t* __restrict__ Bt,
    int M, int N, int K, int mode,
    const float* __restrict__ rcos, const float* __restrict__ rsin,
    bf16_t* __restrict__ outb, float* __restrict__ outf) {
  __shared__ alignas(32) bf16_t As[2][128][32];
  __shared__ alignas(32) bf16_t Bs[2][128][32];

  const int tid  = threadIdx.x;
  const int lane = tid & 31;
  const int wave = tid >> 5;
  const int wm   = wave & 3;          // wave row: 32 rows each
  const int wn   = wave >> 2;         // wave col: 64 cols each
  const int lr   = lane & 15;
  const int lh   = lane >> 4;
  const int m0   = blockIdx.y * 128;
  const int n0   = blockIdx.x * 128;

  // each thread DMAs 2x16B of A-tile and 2x16B of B-tile per stage
  auto stage = [&](int kt, int buf) {
    const int k0 = kt << 5;
#pragma unroll
    for (int i = 0; i < 2; ++i) {
      int lin = tid + i * 256;        // 0..511 -> 128 rows x 4 segs of 8 bf16
      int row = lin >> 2;
      int seg = lin & 3;
      async_b128(lds_off_of(&As[buf][row][seg * 8]),
                 A + (size_t)(m0 + row) * K + k0 + seg * 8);
      async_b128(lds_off_of(&Bs[buf][row][seg * 8]),
                 Bt + (size_t)(n0 + row) * K + k0 + seg * 8);
    }
  };

  const v8f vzero = {0.f, 0.f, 0.f, 0.f, 0.f, 0.f, 0.f, 0.f};
  v8f acc[2][4];
#pragma unroll
  for (int i = 0; i < 2; ++i)
#pragma unroll
    for (int j = 0; j < 4; ++j) acc[i][j] = vzero;

  const int ktiles = K >> 5;
  stage(0, 0);
  wait_async0();
  __syncthreads();

  for (int kt = 0; kt < ktiles; ++kt) {
    const int buf = kt & 1;
    if (kt + 1 < ktiles) stage(kt + 1, buf ^ 1);   // overlap next copy

    // ---- 2x4 WMMA tiles per wave on the current buffer ----
    v16bf af[2], bfrag[4];
#pragma unroll
    for (int mi = 0; mi < 2; ++mi)
      af[mi] = *(const v16bf*)(&As[buf][wm * 32 + mi * 16 + lr][lh * 16]);
#pragma unroll
    for (int ni = 0; ni < 4; ++ni)
      bfrag[ni] = *(const v16bf*)(&Bs[buf][wn * 64 + ni * 16 + lr][lh * 16]);
#pragma unroll
    for (int mi = 0; mi < 2; ++mi)
#pragma unroll
      for (int ni = 0; ni < 4; ++ni)
        acc[mi][ni] = wmma_bf16(af[mi], bfrag[ni], acc[mi][ni]);

    if (kt + 1 < ktiles) wait_async0();
    __syncthreads();
  }

  // ---- epilogue; C layout: lane 0-15 -> rows r, lane 16-31 -> rows r+8 ----
#pragma unroll
  for (int mi = 0; mi < 2; ++mi)
#pragma unroll
    for (int ni = 0; ni < 4; ++ni)
#pragma unroll
      for (int r = 0; r < 8; ++r) {
        const int row = m0 + wm * 32 + mi * 16 + r + 8 * lh;  // token index
        const int col = n0 + wn * 64 + ni * 16 + lr;          // feature index
        const float val = acc[mi][ni][r];
        if (mode == 3) {
          outf[(size_t)row * N + col] = val;
        } else {
          const int t  = row & 2047;      // T = 2048
          const int bb = row >> 11;
          const int hh = col >> 7;        // HD = 128
          const int d  = col & 127;
          if (mode == 2) {
            // V stored transposed: [B,NH,HD,T]
            outb[(((size_t)bb * 16 + hh) * 128 + d) * 2048 + t] = to_bf16(val);
          } else {
            // RoPE: pairs (even d, odd d) live in adjacent lanes (wave32)
            const float c = rcos[t * 64 + (d >> 1)];
            const float s = rsin[t * 64 + (d >> 1)];
            const float partner = __shfl_xor(val, 1);
            const float o = ((lr & 1) == 0) ? (val * c - partner * s)
                                            : (partner * s + val * c);
            // Q/K stored [B,NH,T,HD]
            outb[(((size_t)bb * 16 + hh) * 2048 + t) * 128 + d] = to_bf16(o);
          }
        }
      }
}

// ---------------------------------------------------------------------------
// Flash attention (no causal mask, matching the reference softmax over all k).
// Block = 128 threads (4 waves); each wave owns a 16-row Q tile; block covers
// 64 q rows of one (b,h). 32-key blocks async-DMA'd into LDS, online softmax.
// ---------------------------------------------------------------------------
__global__ __launch_bounds__(128) void flash_attn_kernel(
    const bf16_t* __restrict__ Q,   // [B,NH,T,HD]
    const bf16_t* __restrict__ Km,  // [B,NH,T,HD]
    const bf16_t* __restrict__ Vt,  // [B,NH,HD,T]
    bf16_t* __restrict__ Oattn) {   // [B,T,NH*HD]
  constexpr int T = 2048, HD = 128, NH = 16;
  __shared__ alignas(32) bf16_t Ks[32][128];   // 32 keys x HD
  __shared__ alignas(32) bf16_t Vs[128][32];   // HD x 32 keys (V^T)
  __shared__ alignas(32) bf16_t Ps[4][16][32]; // per-wave P reshape buffer

  const int tid  = threadIdx.x;
  const int lane = tid & 31;
  const int wave = tid >> 5;
  const int lr   = lane & 15;
  const int lh   = lane >> 4;
  const int bh = blockIdx.y;
  const int b = bh >> 4, h = bh & 15;
  const int q0 = blockIdx.x * 64 + wave * 16;

  const bf16_t* Qh = Q  + ((size_t)(b * NH + h) * T) * HD;
  const bf16_t* Kh = Km + ((size_t)(b * NH + h) * T) * HD;
  const bf16_t* Vh = Vt + ((size_t)(b * NH + h) * HD) * T;

  // Q fragments for this wave's 16 rows: 4 chunks of 32 over HD
  v16bf qf[4];
#pragma unroll
  for (int kq = 0; kq < 4; ++kq)
    qf[kq] = *(const v16bf*)(Qh + (size_t)(q0 + lr) * HD + kq * 32 + lh * 16);

  const v8f vzero = {0.f, 0.f, 0.f, 0.f, 0.f, 0.f, 0.f, 0.f};
  v8f accO[8];
  float mi[8], li[8];
#pragma unroll
  for (int i = 0; i < 8; ++i) { accO[i] = vzero; mi[i] = -1e30f; li[i] = 0.f; }
  const float scale = 0.08838834764831845f;  // 1/sqrt(128)

  for (int kb = 0; kb < T / 32; ++kb) {
    // ---- async-DMA 32 keys of K and V^T into LDS ----
#pragma unroll
    for (int i = 0; i < 4; ++i) {
      int lin = tid + i * 128;  // 0..511
      int kr = lin >> 4, ks = lin & 15;
      async_b128(lds_off_of(&Ks[kr][ks * 8]),
                 Kh + (size_t)(kb * 32 + kr) * HD + ks * 8);
      int vr = lin >> 2, vs = lin & 3;
      async_b128(lds_off_of(&Vs[vr][vs * 8]),
                 Vh + (size_t)vr * T + kb * 32 + vs * 8);
    }
    if (kb + 1 < T / 32) {      // pull next block toward L2 while we compute
      __builtin_prefetch(Kh + (size_t)((kb + 1) * 32) * HD, 0, 3);
      __builtin_prefetch(Vh + (kb + 1) * 32, 0, 3);
    }
    wait_async0();
    __syncthreads();

    // ---- S = (Q K^T) * scale : two 16x16 tiles over 32 keys ----
    v8f s0 = vzero, s1 = vzero;
#pragma unroll
    for (int kq = 0; kq < 4; ++kq) {
      v16bf k0f = *(const v16bf*)(&Ks[lr][kq * 32 + lh * 16]);
      v16bf k1f = *(const v16bf*)(&Ks[16 + lr][kq * 32 + lh * 16]);
      s0 = wmma_bf16(qf[kq], k0f, s0);
      s1 = wmma_bf16(qf[kq], k1f, s1);
    }

    // ---- online softmax (fp32). Row r lives in lanes sharing lane-half;
    //      reduce across 16 lanes with xor 1,2,4,8 (wave32). ----
    float p0[8], p1[8], alpha[8];
#pragma unroll
    for (int r = 0; r < 8; ++r) {
      float a = s0[r] * scale;
      float c = s1[r] * scale;
      float lm = fmaxf(a, c);
      lm = fmaxf(lm, __shfl_xor(lm, 1));
      lm = fmaxf(lm, __shfl_xor(lm, 2));
      lm = fmaxf(lm, __shfl_xor(lm, 4));
      lm = fmaxf(lm, __shfl_xor(lm, 8));
      float mn = fmaxf(mi[r], lm);
      alpha[r] = __expf(mi[r] - mn);
      mi[r] = mn;
      p0[r] = __expf(a - mn);
      p1[r] = __expf(c - mn);
      float rs = p0[r] + p1[r];
      rs += __shfl_xor(rs, 1);
      rs += __shfl_xor(rs, 2);
      rs += __shfl_xor(rs, 4);
      rs += __shfl_xor(rs, 8);
      li[r] = li[r] * alpha[r] + rs;
    }
#pragma unroll
    for (int nh = 0; nh < 8; ++nh)
#pragma unroll
      for (int r = 0; r < 8; ++r) accO[nh][r] *= alpha[r];

    // ---- reshape P (C layout -> A fragment layout) through LDS ----
#pragma unroll
    for (int r = 0; r < 8; ++r) {
      Ps[wave][r + 8 * lh][lr]      = to_bf16(p0[r]);
      Ps[wave][r + 8 * lh][16 + lr] = to_bf16(p1[r]);
    }
    __syncthreads();
    v16bf pf = *(const v16bf*)(&Ps[wave][lr][lh * 16]);

    // ---- O += P * V : 8 HD subtiles of 16 ----
#pragma unroll
    for (int nh = 0; nh < 8; ++nh) {
      v16bf vf = *(const v16bf*)(&Vs[nh * 16 + lr][lh * 16]);
      accO[nh] = wmma_bf16(pf, vf, accO[nh]);
    }
    __syncthreads();  // protect Ks/Vs before next staging
  }

  // ---- normalize and store [B,T,NH*HD] bf16 ----
#pragma unroll
  for (int nh = 0; nh < 8; ++nh)
#pragma unroll
    for (int r = 0; r < 8; ++r) {
      const int row = r + 8 * lh;
      const float o = accO[nh][r] / li[r];
      const size_t idx =
          ((size_t)b * T + q0 + row) * (NH * HD) + h * HD + nh * 16 + lr;
      Oattn[idx] = to_bf16(o);
    }
}

// ---------------------------------------------------------------------------
extern "C" void kernel_launch(void* const* d_in, const int* in_sizes, int n_in,
                              void* d_out, int out_size, void* d_ws,
                              size_t ws_size, hipStream_t stream) {
  (void)in_sizes; (void)n_in; (void)out_size; (void)ws_size;
  const float* x    = (const float*)d_in[0];
  const float* rcos = (const float*)d_in[1];
  const float* rsin = (const float*)d_in[2];
  const float* Wq   = (const float*)d_in[3];
  const float* Wk   = (const float*)d_in[4];
  const float* Wv   = (const float*)d_in[5];
  const float* Wo   = (const float*)d_in[6];
  float* out = (float*)d_out;

  const size_t MT = 4096;  // B*T
  const size_t D  = 2048;

  char* ws = (char*)d_ws;
  size_t off = 0;
  auto take = [&](size_t bytes) -> char* {
    char* p = ws + off;
    off = (off + bytes + 255) & ~(size_t)255;
    return p;
  };
  bf16_t* Xb   = (bf16_t*)take(MT * D * 2);      // x, bf16 [4096,2048]
  bf16_t* WqT  = (bf16_t*)take(D * D * 2);       // Wq^T bf16
  bf16_t* WkT  = (bf16_t*)take(D * D * 2);
  bf16_t* WvT  = (bf16_t*)take(D * D * 2);
  bf16_t* WoT  = (bf16_t*)take(D * D * 2);
  bf16_t* Qb   = (bf16_t*)take(MT * D * 2);      // [B,NH,T,HD]
  bf16_t* Kb   = (bf16_t*)take(MT * D * 2);      // [B,NH,T,HD]
  bf16_t* Vt   = (bf16_t*)take(MT * D * 2);      // [B,NH,HD,T]
  bf16_t* Attn = (bf16_t*)take(MT * D * 2);      // [B,T,NH*HD]

  // 1) cast / transpose-cast inputs to bf16
  cast_bf16_kernel<<<4096, 256, 0, stream>>>(x, Xb, MT * D);
  const int tc_grid = (int)((D * D + 255) / 256);
  transpose_cast_kernel<<<tc_grid, 256, 0, stream>>>(Wq, WqT, (int)D, (int)D);
  transpose_cast_kernel<<<tc_grid, 256, 0, stream>>>(Wk, WkT, (int)D, (int)D);
  transpose_cast_kernel<<<tc_grid, 256, 0, stream>>>(Wv, WvT, (int)D, (int)D);
  transpose_cast_kernel<<<tc_grid, 256, 0, stream>>>(Wo, WoT, (int)D, (int)D);

  // 2) QKV projections with fused RoPE / transposed-V epilogues
  dim3 ggrid(16, 32);  // N/128, M/128
  gemm_bf16_kernel<<<ggrid, 256, 0, stream>>>(Xb, WqT, (int)MT, (int)D, (int)D,
                                              0, rcos, rsin, Qb, nullptr);
  gemm_bf16_kernel<<<ggrid, 256, 0, stream>>>(Xb, WkT, (int)MT, (int)D, (int)D,
                                              1, rcos, rsin, Kb, nullptr);
  gemm_bf16_kernel<<<ggrid, 256, 0, stream>>>(Xb, WvT, (int)MT, (int)D, (int)D,
                                              2, rcos, rsin, Vt, nullptr);

  // 3) attention (full softmax, flash-style)
  flash_attn_kernel<<<dim3(32, 32), 128, 0, stream>>>(Qb, Kb, Vt, Attn);

  // 4) output projection, fp32 store to d_out
  gemm_bf16_kernel<<<ggrid, 256, 0, stream>>>(Attn, WoT, (int)MT, (int)D,
                                              (int)D, 3, nullptr, nullptr,
                                              nullptr, out);
}
